// SelfAttentionBlock_16776142258212
// MI455X (gfx1250) — matrix operand order repaired and checked by reading, not compiled, not run on hardware
//
#include <hip/hip_runtime.h>

// Problem constants (from reference): x [8, 512, 32, 32], H=8 heads, d=64,
// s = 32*32 = 1024 tokens/batch, 8192 tokens total, GroupNorm(32, 512).
#define BATCH 8
#define CCH   512
#define SPA   1024
#define NHEAD 8
#define HD    64
#define NTOK  (BATCH * SPA)   // 8192
#define WELEM (CCH * CCH)     // 262144

#if __has_builtin(__builtin_amdgcn_tensor_load_to_lds)
#define HAVE_TDM 1
#else
#define HAVE_TDM 0
#endif

typedef __attribute__((ext_vector_type(16))) _Float16 v16h;
typedef __attribute__((ext_vector_type(8)))  _Float16 v8h;
typedef __attribute__((ext_vector_type(8)))  float    v8f;
typedef __attribute__((ext_vector_type(4)))  unsigned v4u;
typedef __attribute__((ext_vector_type(8)))  int      v8i;
typedef __attribute__((ext_vector_type(4)))  int      v4i;

#define DEV static __device__ __forceinline__

DEV v8f wmma16(v16h a, v16h b, v8f c) {
  return __builtin_amdgcn_wmma_f32_16x16x32_f16(
      /*neg_a=*/false, a, /*neg_b=*/false, b,
      /*c_mod=*/(short)0, c, /*reuse_a=*/false, /*reuse_b=*/false);
}

// A fragment: 16(M) x 32(K) f16 tile from a row-major [rows x stride] matrix.
// ISA layout: lane -> M = lane&15; half = lane>>4.
//   halves 0..7  -> K = k0 + 8*half + (0..7)        (contiguous 16 B)
//   halves 8..15 -> K = k0 + 16 + 8*half + (0..7)   (contiguous 16 B)
DEV v16h frag_a(const _Float16* base, int row0, int stride, int k0, int lane) {
  int m = row0 + (lane & 15);
  int half = (lane >> 4) & 1;
  const _Float16* p = base + m * stride + k0 + 8 * half;
  v8h lo = *(const v8h*)p;
  v8h hi = *(const v8h*)(p + 16);
  v16h r;
#pragma unroll
  for (int i = 0; i < 8; ++i) { r[i] = lo[i]; r[8 + i] = hi[i]; }
  return r;
}

// B fragment: 32(K) x 16(N) f16 tile where B[k][n] = src[n0+n][k0+k] for a
// row-major [N x stride] source (covers W [out,in] for y=xW^T, K rows for
// S=QK^T, and V^T for P*V).
// ISA layout: lane -> N = lane&15; halves 0..7 -> K = k0 + 8*(lane>>4) + h,
// halves 8..15 -> K = k0 + 16 + 8*(lane>>4) + (h-8) (two 16-B loads).
DEV v16h frag_b(const _Float16* base, int n0, int stride, int k0, int lane) {
  int n = n0 + (lane & 15);
  int half = (lane >> 4) & 1;
  const _Float16* p = base + n * stride + k0 + 8 * half;
  v8h lo = *(const v8h*)p;
  v8h hi = *(const v8h*)(p + 16);
  v16h r;
#pragma unroll
  for (int i = 0; i < 8; ++i) { r[i] = lo[i]; r[8 + i] = hi[i]; }
  return r;
}

// Truncating a flat pointer to a __shared__ object yields the LDS byte offset
// (flat LDS addressing keeps the offset in addr[31:0]).
DEV unsigned lds_off(const void* p) { return (unsigned)(unsigned long long)p; }

#if HAVE_TDM
// TDM 2D tile load global->LDS (one descriptor, issued by a single wave,
// tracked by TENSORcnt).  D# per CDNA5 ISA §8.3/8.4:
//   group0: count=1 | lds_addr[63:32] | global_addr[120:64] | type=2[127:126]
//   group1: data_size=2B | tensor_dim0/1 | tile_dim0/1 | tensor_dim0_stride
// tile_d0 = elements per row, tile_d1 = rows; stride0 = row pitch (elements).
// This toolchain's builtin takes 6 args: (g0, g1, g2, g3, g_extra, cpol).
DEV void tdm_load_2d(unsigned lds, const void* g, unsigned tile_d0,
                     unsigned tile_d1, unsigned long long stride0) {
  unsigned long long ga = (unsigned long long)g;
  v4u g0 = { 1u,                                   // count=1, is_restore=0
             lds,                                  // lds_addr (bytes)
             (unsigned)ga,                         // global_addr[31:0]
             (unsigned)(ga >> 32) | 0x80000000u }; // addr[56:32] | type=2
  unsigned td0 = tile_d0, td1 = tile_d1;           // tensor dims == tile dims
  v8i g1 = { (int)(1u << 16),                      // data_size=1 -> 2 bytes
             (int)((td0 & 0xFFFFu) << 16),         // tensor_dim0[15:0]
             (int)((td0 >> 16) | ((td1 & 0xFFFFu) << 16)),
             (int)((td1 >> 16) | (tile_d0 << 16)), // tile_dim0
             (int)tile_d1,                         // tile_dim1 (tile_dim2=0)
             (int)(unsigned)stride0,               // dim0_stride[31:0]
             (int)(unsigned)(stride0 >> 32),       // dim0_stride[47:32]
             0 };
  v4i z4 = { 0, 0, 0, 0 };                         // groups 2/3 unused (2D)
  v8i z8 = { 0, 0, 0, 0, 0, 0, 0, 0 };
  __builtin_amdgcn_tensor_load_to_lds(g0, g1, z4, z4, z8, 0);
}
#else
// Async copy 16 B/lane global -> LDS, tracked by ASYNCcnt.
DEV void async_b128(unsigned lds, const _Float16* g) {
  unsigned long long ga = (unsigned long long)g;
  asm volatile("global_load_async_to_lds_b128 %0, %1, off"
               :: "v"(lds), "v"(ga) : "memory");
}
#endif

// ---------------------------------------------------------------------------
// Kernel 1: convert the four f32 weight matrices to f16, packed [Wq|Wk|Wv|Wo].
// ---------------------------------------------------------------------------
__global__ __launch_bounds__(256) void cvt_w_kernel(
    const float* __restrict__ wq, const float* __restrict__ wk,
    const float* __restrict__ wv, const float* __restrict__ wo,
    _Float16* __restrict__ w16) {
  int e = blockIdx.x * 256 + threadIdx.x;           // 0 .. 4*WELEM-1
  int mat = e >> 18;                                 // /WELEM
  int off = e & (WELEM - 1);
  const float* src = (mat == 0) ? wq : (mat == 1) ? wk : (mat == 2) ? wv : wo;
  w16[e] = (_Float16)src[off];
}

// ---------------------------------------------------------------------------
// Kernel 2: GroupNorm(32, 512) over x [8, 512, 32, 32] -> token-major f16
// activations t16 [8192, 512].  One block per (batch, group): 16 ch x 1024 hw.
// ---------------------------------------------------------------------------
__global__ __launch_bounds__(256) void gn_kernel(
    const float* __restrict__ x, const float* __restrict__ sc,
    const float* __restrict__ bi, _Float16* __restrict__ t16) {
  __shared__ float s1[256];
  __shared__ float s2[256];
  int b = blockIdx.x >> 5;
  int g = blockIdx.x & 31;
  const float* xg = x + (b * CCH + g * 16) * SPA;    // 16 channels * 1024
  int tid = threadIdx.x;
  float a = 0.f, q = 0.f;
#pragma unroll 4
  for (int i = tid; i < 16 * SPA; i += 256) {
    float v = xg[i];
    a += v; q += v * v;
  }
  s1[tid] = a; s2[tid] = q;
  __syncthreads();
  for (int off = 128; off > 0; off >>= 1) {
    if (tid < off) { s1[tid] += s1[tid + off]; s2[tid] += s2[tid + off]; }
    __syncthreads();
  }
  float mean = s1[0] * (1.f / 16384.f);
  float var  = s2[0] * (1.f / 16384.f) - mean * mean;
  float rs   = rsqrtf(var + 1e-5f);
  for (int i = tid; i < 16 * SPA; i += 256) {
    int c  = g * 16 + (i >> 10);
    int hw = i & (SPA - 1);
    float v = (xg[i] - mean) * rs * sc[c] + bi[c];
    t16[(b * SPA + hw) * CCH + c] = (_Float16)v;     // token-major
  }
}

// ---------------------------------------------------------------------------
// Kernel 3: QKV projection.  t16 [8192,512] x W^T -> q,k token-major f16
// [8192,512] and v transposed vT [b*512, 1024].  One wave = 32x64 output tile
// (8 accumulators): 8 WMMAs per 12 b128 loads per K-step.
// ---------------------------------------------------------------------------
__global__ __launch_bounds__(256) void qkv_kernel(
    const _Float16* __restrict__ t16, const _Float16* __restrict__ w16,
    const float* __restrict__ bq, const float* __restrict__ bk,
    const float* __restrict__ bv,
    _Float16* __restrict__ qb, _Float16* __restrict__ kb,
    _Float16* __restrict__ vT) {
  int lane = threadIdx.x & 31;
  int wave = threadIdx.x >> 5;
  int half = lane >> 4;
  int tile = blockIdx.x * 8 + wave;       // 0 .. 3*256*8-1
  int mat  = tile >> 11;                  // /2048 : 0=q, 1=k, 2=v
  int rem  = tile & 2047;
  int m0   = (rem >> 3) * 32;             // token tile
  int n0   = (rem & 7) * 64;              // channel tile
  const _Float16* w = w16 + mat * WELEM;

  v8f acc[2][4] = {};
  for (int k0 = 0; k0 < CCH; k0 += 32) {
    v16h a0 = frag_a(t16, m0,      CCH, k0, lane);
    v16h a1 = frag_a(t16, m0 + 16, CCH, k0, lane);
#pragma unroll
    for (int j = 0; j < 4; ++j) {
      v16h bj = frag_b(w, n0 + j * 16, CCH, k0, lane);
      acc[0][j] = wmma16(a0, bj, acc[0][j]);
      acc[1][j] = wmma16(a1, bj, acc[1][j]);
    }
  }

  const float* bias = (mat == 0) ? bq : (mat == 1) ? bk : bv;
#pragma unroll
  for (int i = 0; i < 2; ++i) {
#pragma unroll
    for (int j = 0; j < 4; ++j) {
      int mt = m0 + i * 16;
      int n  = n0 + j * 16 + (lane & 15);
      float bn = bias[n];
      if (mat < 2) {
        _Float16* dst = (mat == 0) ? qb : kb;
#pragma unroll
        for (int r = 0; r < 8; ++r) {
          int m = mt + r + 8 * half;      // D layout: VGPR r -> M = r+8*half
          dst[m * CCH + n] = (_Float16)(acc[i][j][r] + bn);
        }
      } else {                            // v: store transposed [b*512+n, s]
        int b  = mt >> 10;
        int s0 = (mt & (SPA - 1)) + 8 * half;
        v8h pk;
#pragma unroll
        for (int r = 0; r < 8; ++r) pk[r] = (_Float16)(acc[i][j][r] + bn);
        *(v8h*)(vT + ((b * CCH + n) * SPA + s0)) = pk;  // 16-B contiguous
      }
    }
  }
}

// ---------------------------------------------------------------------------
// Kernel 4: flash attention per (b,h).  All 8 waves of a block share one
// (b,h): each 32-key K-chunk (32x64 f16) and V-chunk (64x32 f16) is staged
// into LDS once per block.  With TDM: wave 0 issues two tensor_load_to_lds
// DMAs per chunk (TENSORcnt); otherwise cooperative
// global_load_async_to_lds_b128 (ASYNCcnt).  Double-buffered either way.
// Each wave owns 16 query rows: 8 WMMAs per chunk with online softmax; P is
// transposed D->A layout through a wave-private LDS bounce (LDS ops are
// in-order within a wave, so no barrier needed there).
// ---------------------------------------------------------------------------
__global__ __launch_bounds__(256) void attn_kernel(
    const _Float16* __restrict__ qb, const _Float16* __restrict__ kb,
    const _Float16* __restrict__ vT, _Float16* __restrict__ ab) {
  __shared__ _Float16 kshm[2][32 * 64];   // [key][d]   4 KB per buffer
  __shared__ _Float16 vshm[2][64 * 32];   // [d][key]   4 KB per buffer
  __shared__ _Float16 psh[8 * 16 * 32];   // per-wave P transpose buffers
  int tid  = threadIdx.x;
  int lane = tid & 31;
  int wave = tid >> 5;
  int half = lane >> 4;
  int b = blockIdx.x >> 3, h = blockIdx.x & 7;
  int qr   = blockIdx.y * 128 + wave * 16;       // query rows within batch
  int qtok = b * SPA + qr;
  int hc   = h * HD;

#if HAVE_TDM
  // Uniform tile-start pointers for the DMA descriptors.
  const _Float16* kg0 = kb + (size_t)(b * SPA) * CCH + hc;     // + kc*CCH
  const _Float16* vg0 = vT + (size_t)(b * CCH + hc) * SPA;     // + kc
  if (wave == 0) {                               // stage chunk 0 into buf 0
    tdm_load_2d(lds_off(kshm[0]), kg0, 64, 32, CCH);
    tdm_load_2d(lds_off(vshm[0]), vg0, 32, 64, SPA);
  }
#else
  // Per-thread staging slice: 16 B each, 256 threads cover 4 KB per buffer.
  int kr = tid >> 3, kco = (tid & 7) * 8;        // K: row=key, col=d
  const _Float16* kgp = kb + (b * SPA + kr) * CCH + hc + kco;
  unsigned klds[2] = { lds_off(&kshm[0][kr * 64 + kco]),
                       lds_off(&kshm[1][kr * 64 + kco]) };
  int vd = tid >> 2, vco = (tid & 3) * 8;        // V^T: row=d, col=key
  const _Float16* vgp = vT + (b * CCH + hc + vd) * SPA + vco;
  unsigned vlds[2] = { lds_off(&vshm[0][vd * 32 + vco]),
                       lds_off(&vshm[1][vd * 32 + vco]) };
  async_b128(klds[0], kgp);
  async_b128(vlds[0], vgp);
#endif

  v16h aq0 = frag_a(qb, qtok, CCH, hc,      lane);
  v16h aq1 = frag_a(qb, qtok, CCH, hc + 32, lane);

  v8f o0 = {}, o1 = {}, o2 = {}, o3 = {};
  float mrow[8], lrow[8];
#pragma unroll
  for (int r = 0; r < 8; ++r) { mrow[r] = -1e30f; lrow[r] = 0.f; }
  _Float16* myp = psh + wave * 512;
  const float scale = 0.125f;                    // 1/sqrt(64)
  const v8f zero = {};

  for (int i = 0; i < 32; ++i) {
    int cur = i & 1;
#if HAVE_TDM
    if (wave == 0) {
      if (i + 1 < 32) {                          // stage next chunk
        int kc = (i + 1) * 32;
        tdm_load_2d(lds_off(kshm[cur ^ 1]), kg0 + (size_t)kc * CCH, 64, 32, CCH);
        tdm_load_2d(lds_off(vshm[cur ^ 1]), vg0 + kc, 32, 64, SPA);
        __builtin_amdgcn_s_wait_tensorcnt((short)2);  // chunk i landed
      } else {
        __builtin_amdgcn_s_wait_tensorcnt((short)0);
      }
    }
#else
    if (i + 1 < 32) {                            // stage next chunk
      int kc = (i + 1) * 32;
      async_b128(klds[cur ^ 1], kgp + kc * CCH);
      async_b128(vlds[cur ^ 1], vgp + kc);
      asm volatile("s_wait_asynccnt 0x2" ::: "memory");  // chunk i landed
    } else {
      asm volatile("s_wait_asynccnt 0x0" ::: "memory");
    }
#endif
    __syncthreads();                             // fills visible to all waves
    const _Float16* kc_s = kshm[cur];
    const _Float16* vc_s = vshm[cur];

    // S = Q K^T (two 16-column tiles, K=64 over two k-steps), from LDS
    v16h bk0 = frag_b(kc_s, 0,  64, 0,  lane);
    v16h bk1 = frag_b(kc_s, 0,  64, 32, lane);
    v8f  sA  = wmma16(aq1, bk1, wmma16(aq0, bk0, zero));
    v16h bk2 = frag_b(kc_s, 16, 64, 0,  lane);
    v16h bk3 = frag_b(kc_s, 16, 64, 32, lane);
    v8f  sB  = wmma16(aq1, bk3, wmma16(aq0, bk2, zero));

    float alpha[8];
    v8f pA, pB;
#pragma unroll
    for (int r = 0; r < 8; ++r) {
      float sa = sA[r] * scale, sb = sB[r] * scale;
      float v = fmaxf(sa, sb);                   // row max across 32 cols
      v = fmaxf(v, __shfl_xor(v, 1));
      v = fmaxf(v, __shfl_xor(v, 2));
      v = fmaxf(v, __shfl_xor(v, 4));
      v = fmaxf(v, __shfl_xor(v, 8));
      float mn = fmaxf(mrow[r], v);
      float al = __expf(mrow[r] - mn);
      float pa = __expf(sa - mn);
      float pb = __expf(sb - mn);
      float rs = pa + pb;
      rs += __shfl_xor(rs, 1);
      rs += __shfl_xor(rs, 2);
      rs += __shfl_xor(rs, 4);
      rs += __shfl_xor(rs, 8);
      lrow[r] = lrow[r] * al + rs;
      mrow[r] = mn;
      alpha[r] = al;
      pA[r] = pa; pB[r] = pb;
    }
#pragma unroll
    for (int r = 0; r < 8; ++r) {
      o0[r] *= alpha[r]; o1[r] *= alpha[r];
      o2[r] *= alpha[r]; o3[r] *= alpha[r];
    }
    // transpose P (D layout -> A layout) via wave-private LDS bounce
    int col = lane & 15;
#pragma unroll
    for (int r = 0; r < 8; ++r) {
      int m = r + 8 * half;
      myp[m * 32 + col]      = (_Float16)pA[r];
      myp[m * 32 + 16 + col] = (_Float16)pB[r];
    }
    v16h ap = frag_a(myp, 0, 32, 0, lane);       // ds_load_b128 x2 per lane
    // O += P * V  (V^T rows are d-channels, keys contiguous), from LDS
    o0 = wmma16(ap, frag_b(vc_s, 0,  32, 0, lane), o0);
    o1 = wmma16(ap, frag_b(vc_s, 16, 32, 0, lane), o1);
    o2 = wmma16(ap, frag_b(vc_s, 32, 32, 0, lane), o2);
    o3 = wmma16(ap, frag_b(vc_s, 48, 32, 0, lane), o3);
    __syncthreads();                             // done reading before refill
  }

#pragma unroll
  for (int r = 0; r < 8; ++r) {
    float inv = 1.0f / lrow[r];
    o0[r] *= inv; o1[r] *= inv; o2[r] *= inv; o3[r] *= inv;
  }
  // store token-major [8192, 512]
#pragma unroll
  for (int j = 0; j < 4; ++j) {
    const v8f& o = (j == 0) ? o0 : (j == 1) ? o1 : (j == 2) ? o2 : o3;
    int n = hc + j * 16 + (lane & 15);
#pragma unroll
    for (int r = 0; r < 8; ++r) {
      int tok = qtok + r + 8 * half;
      ab[tok * CCH + n] = (_Float16)o[r];
    }
  }
}

// ---------------------------------------------------------------------------
// Kernel 5: output projection + bias + residual, f32 output in [b,c,h,w].
// One wave = 32x64 tile; per-lane rows are contiguous in hw -> float4 I/O.
// ---------------------------------------------------------------------------
__global__ __launch_bounds__(256) void proj_kernel(
    const _Float16* __restrict__ ab, const _Float16* __restrict__ wo16,
    const float* __restrict__ bo, const float* __restrict__ x,
    float* __restrict__ out) {
  int lane = threadIdx.x & 31;
  int wave = threadIdx.x >> 5;
  int half = lane >> 4;
  int tile = blockIdx.x * 8 + wave;       // 0 .. 2047
  int m0 = (tile >> 3) * 32;
  int n0 = (tile & 7) * 64;

  v8f acc[2][4] = {};
  for (int k0 = 0; k0 < CCH; k0 += 32) {
    v16h a0 = frag_a(ab, m0,      CCH, k0, lane);
    v16h a1 = frag_a(ab, m0 + 16, CCH, k0, lane);
#pragma unroll
    for (int j = 0; j < 4; ++j) {
      v16h bj = frag_b(wo16, n0 + j * 16, CCH, k0, lane);
      acc[0][j] = wmma16(a0, bj, acc[0][j]);
      acc[1][j] = wmma16(a1, bj, acc[1][j]);
    }
  }
#pragma unroll
  for (int i = 0; i < 2; ++i) {
#pragma unroll
    for (int j = 0; j < 4; ++j) {
      int mt = m0 + i * 16;
      int n  = n0 + j * 16 + (lane & 15);
      float bn = bo[n];
      int mb = mt + 8 * half;
      int b  = mb >> 10;
      int hw = mb & (SPA - 1);
      const float* xr = x   + ((b * CCH + n) * SPA + hw);
      float*       op = out + ((b * CCH + n) * SPA + hw);
      float4 xlo = *(const float4*)xr;
      float4 xhi = *(const float4*)(xr + 4);
      float4 olo, ohi;
      olo.x = acc[i][j][0] + bn + xlo.x;
      olo.y = acc[i][j][1] + bn + xlo.y;
      olo.z = acc[i][j][2] + bn + xlo.z;
      olo.w = acc[i][j][3] + bn + xlo.w;
      ohi.x = acc[i][j][4] + bn + xhi.x;
      ohi.y = acc[i][j][5] + bn + xhi.y;
      ohi.z = acc[i][j][6] + bn + xhi.z;
      ohi.w = acc[i][j][7] + bn + xhi.w;
      *(float4*)op       = olo;
      *(float4*)(op + 4) = ohi;
    }
  }
}

// ---------------------------------------------------------------------------
extern "C" void kernel_launch(void* const* d_in, const int* in_sizes, int n_in,
                              void* d_out, int out_size, void* d_ws,
                              size_t ws_size, hipStream_t stream) {
  const float* x  = (const float*)d_in[0];
  const float* gs = (const float*)d_in[1];
  const float* gb = (const float*)d_in[2];
  const float* Wq = (const float*)d_in[3];
  const float* bq = (const float*)d_in[4];
  const float* Wk = (const float*)d_in[5];
  const float* bk = (const float*)d_in[6];
  const float* Wv = (const float*)d_in[7];
  const float* bv = (const float*)d_in[8];
  const float* Wo = (const float*)d_in[9];
  const float* bo = (const float*)d_in[10];
  float* out = (float*)d_out;

  // Workspace layout (f16 halves); total = 5*4194304 + 4*262144 = 22,020,096
  // halves = ~42 MB.  All buffers fully (re)written every call.
  _Float16* t16 = (_Float16*)d_ws;                 // [8192, 512] normed tokens
  _Float16* w16 = t16 + (size_t)NTOK * CCH;        // [4][512][512] f16 weights
  _Float16* qb  = w16 + (size_t)4 * WELEM;         // [8192, 512]
  _Float16* kb  = qb  + (size_t)NTOK * CCH;        // [8192, 512]
  _Float16* vT  = kb  + (size_t)NTOK * CCH;        // [8*512, 1024] (V^T)
  _Float16* abuf = vT + (size_t)NTOK * CCH;        // [8192, 512] attn out

  cvt_w_kernel<<<4 * WELEM / 256, 256, 0, stream>>>(Wq, Wk, Wv, Wo, w16);
  gn_kernel<<<BATCH * 32, 256, 0, stream>>>(x, gs, gb, t16);
  qkv_kernel<<<3 * (NTOK / 32) * (CCH / 64) / 8, 256, 0, stream>>>(
      t16, w16, bq, bk, bv, qb, kb, vT);
  attn_kernel<<<dim3(BATCH * NHEAD, SPA / 128), 256, 0, stream>>>(
      qb, kb, vT, abuf);
  proj_kernel<<<(NTOK / 32) * (CCH / 64) / 8, 256, 0, stream>>>(
      abuf, w16 + (size_t)3 * WELEM, bo, x, out);
}
